// Classifier_35287451304827
// MI455X (gfx1250) — matrix operand order repaired and checked by reading, not compiled, not run on hardware
//
#include <hip/hip_runtime.h>
#include <hip/hip_bf16.h>
#include <float.h>

typedef __attribute__((ext_vector_type(16))) __bf16 v16bf;
typedef __attribute__((ext_vector_type(8)))  float  v8f;

#define C_FEAT 512
#define H_FEAT 37
#define W_FEAT 50
#define NBOX   128
#define D_FC   25088   /* 512*7*7 */
#define HID    4096
#define NCLS   21

// round-to-nearest-even fp32 -> bf16, two at a time, packed into one dword
// (a in [15:0], b in [31:16]).  The merge of the two rounded high halves is
// a byte permute of {b_rounded, a_rounded}: result bytes = [a.b2, a.b3,
// b.b2, b.b3] -> v_perm_b32 selector 0x07060302 (S0=high dword=b, S1=low=a).
__device__ __forceinline__ unsigned int pack_bf16x2(float a, float b) {
  unsigned int ua = __float_as_uint(a);
  unsigned int ub = __float_as_uint(b);
  ua += 0x7FFFu + ((ua >> 16) & 1u);
  ub += 0x7FFFu + ((ub >> 16) & 1u);
  return __builtin_amdgcn_perm(ub, ua, 0x07060302u);
}

// ---------------------------------------------------------------------------
// Kernel 1: ROI max pooling. rois[n][c*49 + i*7 + j], n<128.
// ---------------------------------------------------------------------------
__global__ void roi_pool_kernel(const float* __restrict__ feat,
                                const int*   __restrict__ boxes,
                                float*       __restrict__ rois) {
  int id = blockIdx.x * blockDim.x + threadIdx.x;
  const int total = NBOX * C_FEAT * 49;
  if (id >= total) return;
  int n   = id / (C_FEAT * 49);
  int rem = id % (C_FEAT * 49);
  int c   = rem / 49;
  int bin = rem % 49;
  int bi  = bin / 7, bj = bin % 7;

  // floor(box * 1/16) == box >> 4 for non-negative int boxes
  int x1 = boxes[n * 4 + 0] >> 4;
  int y1 = boxes[n * 4 + 1] >> 4;
  int x2 = boxes[n * 4 + 2] >> 4;
  int y2 = boxes[n * 4 + 3] >> 4;
  x1 = min(max(x1, 0), W_FEAT - 1);
  y1 = min(max(y1, 0), H_FEAT - 1);
  x2 = min(max(x2, 0), W_FEAT - 1);
  y2 = min(max(y2, 0), H_FEAT - 1);
  int h = y2 - y1 + 1;
  int w = x2 - x1 + 1;

  int rlo = y1 + (bi * h) / 7;
  int rhi = y1 + ((bi + 1) * h + 6) / 7;      // ceil
  int clo = x1 + (bj * w) / 7;
  int chi = x1 + ((bj + 1) * w + 6) / 7;      // ceil

  const float* f = feat + (size_t)c * (H_FEAT * W_FEAT);
  float m = -FLT_MAX;
  for (int r = rlo; r < rhi; ++r)
    for (int cc = clo; cc < chi; ++cc)
      m = fmaxf(m, f[r * W_FEAT + cc]);

  rois[(size_t)n * D_FC + c * 49 + bi * 7 + bj] = m;
}

// ---------------------------------------------------------------------------
// Kernel 2: Y[128 x N] = relu(X[128 x K] * W[K x N] + b), bf16 WMMA, f32 acc.
// Block = 256 threads = 8 waves. Block tile: M=128, N=128, K-step 32.
// Wave w owns N-strip [w*16, w*16+16), all 8 M-tiles (acc in 64 VGPRs).
// Weights stream from HBM exactly once (grid covers N); activations are
// 12.8 MB and live in the 192 MB L2 across the 32 blocks.
// ---------------------------------------------------------------------------
__global__ __launch_bounds__(256)
void fc_wmma_kernel(const float* __restrict__ X, const float* __restrict__ W,
                    const float* __restrict__ bias, float* __restrict__ Y,
                    int K, int N, int relu) {
  // A staged as bf16 [m][k] (128x32), B staged transposed bf16 [n][k] (128x32)
  __shared__ __align__(16) unsigned short As[128 * 32];
  __shared__ __align__(16) unsigned short Bs[128 * 32];

  const int tid  = threadIdx.x;
  const int wv   = tid >> 5;         // wave id 0..7
  const int lane = tid & 31;
  const int nl   = lane & 15;        // N (and M) position within tile
  const int hf   = lane >> 4;        // lane half selects K / M range

  const int colbase = blockIdx.x * 128;

  v8f acc[8];
  v8f zero = {};
  #pragma unroll
  for (int t = 0; t < 8; ++t) acc[t] = zero;

  for (int kk = 0; kk < K; kk += 32) {
    // ---- stage A: 128x32 fp32 -> bf16 LDS -------------------------------
    // thread owns quads of 4 consecutive k: one global_load_b128 +
    // two v_perm packs + one ds_store_b64 per quad.
    #pragma unroll
    for (int i = 0; i < 4; ++i) {
      int q = i * 256 + tid;         // quad id, 1024 quads total
      int m = q >> 3;                // row 0..127
      int k = (q & 7) * 4;           // k 0,4,...,28
      float4 x4 = *(const float4*)(X + (size_t)m * K + kk + k);
      unsigned int lo = pack_bf16x2(x4.x, x4.y);
      unsigned int hi = pack_bf16x2(x4.z, x4.w);
      *(uint2*)&As[m * 32 + k] = make_uint2(lo, hi);
    }
    // ---- stage B: 32x128 fp32 -> bf16 LDS, transposed [n][k] ------------
    // thread owns 4 consecutive k for one n; lanes span consecutive n so
    // each of the 4 loads is coalesced across the wave.
    #pragma unroll
    for (int i = 0; i < 4; ++i) {
      int q = i * 256 + tid;
      int n = q & 127;
      int k = (q >> 7) * 4;          // k 0,4,...,28
      const float* wp = W + (size_t)(kk + k) * N + colbase + n;
      float w0 = wp[0];
      float w1 = wp[(size_t)N];
      float w2 = wp[(size_t)2 * N];
      float w3 = wp[(size_t)3 * N];
      unsigned int lo = pack_bf16x2(w0, w1);
      unsigned int hi = pack_bf16x2(w2, w3);
      *(uint2*)&Bs[n * 32 + k] = make_uint2(lo, hi);
    }
    // pull the next weight tile toward L2 while we compute this one
    if (kk + 32 < K && tid < 32) {
      __builtin_prefetch(&W[(size_t)(kk + 32 + tid) * N + colbase], 0, 1);
    }
    __syncthreads();

    const unsigned int* As32 = (const unsigned int*)As;
    const unsigned int* Bs32 = (const unsigned int*)Bs;

    // B fragment (32x16 K x N): lane nl = N, hf selects K half; VGPR v holds
    // K = hf*16 + 2v, 2v+1  -> dword index (wv*16+nl)*16 + hf*8 + v
    union { unsigned int u[8]; v16bf v; } fb;
    #pragma unroll
    for (int v = 0; v < 8; ++v)
      fb.u[v] = Bs32[(wv * 16 + nl) * 16 + hf * 8 + v];

    #pragma unroll
    for (int t = 0; t < 8; ++t) {
      // A fragment (16x32 M x K): lane nl = M; VGPR v holds
      // K = hf*16 + 2v, +1
      union { unsigned int u[8]; v16bf v; } fa;
      #pragma unroll
      for (int v = 0; v < 8; ++v) {
        int ka = hf * 16 + v * 2;
        fa.u[v] = As32[(t * 16 + nl) * 16 + (ka >> 1)];
      }
      acc[t] = __builtin_amdgcn_wmma_f32_16x16x32_bf16(
          false, fa.v, false, fb.v, (short)0, acc[t], false, false);
    }
    __syncthreads();
  }

  // ---- epilogue: C/D layout lane nl = N, VGPR v -> M = hf*8 + v ----
  const int col = colbase + wv * 16 + nl;
  const float bv = bias[col];
  #pragma unroll
  for (int t = 0; t < 8; ++t) {
    #pragma unroll
    for (int v = 0; v < 8; ++v) {
      int row = t * 16 + hf * 8 + v;
      float y = acc[t][v] + bv;
      if (relu) y = fmaxf(y, 0.0f);
      Y[(size_t)row * N + col] = y;
    }
  }
}

// ---------------------------------------------------------------------------
// Kernel 3: cls/reg heads. out = [pred_label 128x21 | deltas 128x84]
// ---------------------------------------------------------------------------
__global__ void heads_kernel(const float* __restrict__ X,
                             const float* __restrict__ cls_w,
                             const float* __restrict__ cls_b,
                             const float* __restrict__ reg_w,
                             const float* __restrict__ reg_b,
                             float* __restrict__ out) {
  int id = blockIdx.x * blockDim.x + threadIdx.x;
  const int total = NBOX * (NCLS + 4 * NCLS);   // 128 * 105
  if (id >= total) return;
  int m = id / 105;
  int j = id % 105;
  const float* x = X + (size_t)m * HID;
  if (j < NCLS) {
    float s = cls_b[j];
    #pragma unroll 4
    for (int k = 0; k < HID; ++k) s = fmaf(x[k], cls_w[(size_t)k * NCLS + j], s);
    out[m * NCLS + j] = s;
  } else {
    int jr = j - NCLS;
    float s = reg_b[jr];
    #pragma unroll 4
    for (int k = 0; k < HID; ++k) s = fmaf(x[k], reg_w[(size_t)k * 84 + jr], s);
    out[NBOX * NCLS + m * 84 + jr] = s;
  }
}

// ---------------------------------------------------------------------------
extern "C" void kernel_launch(void* const* d_in, const int* in_sizes, int n_in,
                              void* d_out, int out_size, void* d_ws, size_t ws_size,
                              hipStream_t stream) {
  const float* feat  = (const float*)d_in[0];
  const int*   boxes = (const int*)  d_in[1];
  const float* fc1_w = (const float*)d_in[2];
  const float* fc1_b = (const float*)d_in[3];
  const float* fc2_w = (const float*)d_in[4];
  const float* fc2_b = (const float*)d_in[5];
  const float* cls_w = (const float*)d_in[6];
  const float* cls_b = (const float*)d_in[7];
  const float* reg_w = (const float*)d_in[8];
  const float* reg_b = (const float*)d_in[9];
  float* out = (float*)d_out;

  char* ws = (char*)d_ws;
  float* rois = (float*)ws;                                    // 128*25088 f32
  float* h1   = (float*)(ws + (size_t)NBOX * D_FC * sizeof(float));
  float* h2   = h1 + (size_t)NBOX * HID;

  int total_pool = NBOX * C_FEAT * 49;
  roi_pool_kernel<<<(total_pool + 255) / 256, 256, 0, stream>>>(feat, boxes, rois);

  fc_wmma_kernel<<<HID / 128, 256, 0, stream>>>(rois, fc1_w, fc1_b, h1, D_FC, HID, 1);
  fc_wmma_kernel<<<HID / 128, 256, 0, stream>>>(h1,   fc2_w, fc2_b, h2, HID,  HID, 1);

  int total_heads = NBOX * 105;
  heads_kernel<<<(total_heads + 255) / 256, 256, 0, stream>>>(h2, cls_w, cls_b,
                                                              reg_w, reg_b, out);
}
